// SAPCABlockV5_39676907888328
// MI455X (gfx1250) — matrix operand order repaired and checked by reading, not compiled, not run on hardware
//
#include <hip/hip_runtime.h>
#include <hip/hip_bf16.h>

// ---------------------------------------------------------------------------
// SA-PCA block for MI455X (gfx1250, wave32, WMMA + async-LDS + TDM).
//   K0  xmean    : per-(b,c) spatial mean of x
//   Kc  f2bf     : conv1 weights fp32 -> bf16
//   Kw2 w2perm   : conv2 weights fp32 -> bf16, tap-major [tap][co][ci]
//   K1  conv1    : g = W1 @ (x - xmean) (bias cancels), bf16 out  [WMMA]
//   K2  cov      : cmat = g g^T / B, bf16 out  [WMMA + async global->LDS]
//   K3  eig      : top-16 eigvecs, 32 rounds subspace iteration; whole 256x256
//                  bf16 covariance TDM-loaded into 320KB WGP LDS  [WMMA + TDM]
//   K4  attn     : proj = W@g, softmax(16), y = W^T z  [WMMA]
//   K5  conv2+BN+res : 3x3 implicit GEMM [WMMA + async], fused BN + residual
// ---------------------------------------------------------------------------

typedef __attribute__((ext_vector_type(16))) __bf16 v16bf;
typedef __attribute__((ext_vector_type(8)))  __bf16 v8bf;
typedef __attribute__((ext_vector_type(8)))  float  v8f;
typedef __attribute__((ext_vector_type(4)))  unsigned int u32x4;
typedef __attribute__((ext_vector_type(4)))  int i32x4;
typedef __attribute__((ext_vector_type(8)))  int i32x8;

// address-space qualified v4i for the async global->LDS builtin
typedef __attribute__((address_space(1))) i32x4 as1_i32x4;
typedef __attribute__((address_space(3))) i32x4 as3_i32x4;

__device__ inline __bf16 to_bf(float x){ return (__bf16)x; }

// --- async global -> LDS (16B per lane), ASYNCcnt tracked -------------------
__device__ inline void async_b128(void* lds, const void* gptr){
#if __has_builtin(__builtin_amdgcn_global_load_async_to_lds_b128)
  __builtin_amdgcn_global_load_async_to_lds_b128(
      (as1_i32x4*)gptr, (as3_i32x4*)lds, 0, 0);
#else
  unsigned loff = (unsigned)(uintptr_t)lds;
  asm volatile("global_load_async_to_lds_b128 %0, %1, off"
               :: "v"(loff), "v"(gptr) : "memory");
#endif
}
__device__ inline void wait_async(){
#if __has_builtin(__builtin_amdgcn_s_wait_asynccnt)
  __builtin_amdgcn_s_wait_asynccnt(0);
#else
  asm volatile("s_wait_asynccnt 0x0" ::: "memory");
#endif
}

// --- WMMA fragment helpers (CDNA5 ISA 7.12.2, wave32) -----------------------
// Both loaders read two aligned 16B LDS chunks -> 2x ds_load_b128 each.
__device__ inline v16bf ld_frag2(const __bf16* p0, const __bf16* p1){
  v8bf a0 = *(const v8bf*)p0;
  v8bf a1 = *(const v8bf*)p1;
  v16bf r;
  #pragma unroll
  for (int i = 0; i < 8; ++i){ r[i] = a0[i]; r[i+8] = a1[i]; }
  return r;
}
// A 16x32: s = [16][ld] tile, k-contiguous rows; elements are runs at
//   k = half*8 + [0..7] and k = 16 + half*8 + [0..7]
__device__ inline v16bf load_A_frag(const __bf16* s, int ld, int lane){
  int m = lane & 15, half = lane >> 4;
  const __bf16* p = s + m*ld + half*8;
  return ld_frag2(p, p + 16);
}
// B 32x16 from transposed tile BsT[n][k] (k contiguous): k = half*8 + [0..7], 16 + half*8 + [0..7]
__device__ inline v16bf load_B_fragT_ld(const __bf16* s, int ld, int lane){
  int n = lane & 15, half = lane >> 4;
  const __bf16* p = s + n*ld + half*8;
  return ld_frag2(p, p + 16);
}
__device__ inline v16bf load_B_fragT(const __bf16* s, int lane){
  return load_B_fragT_ld(s, 32, lane);
}
__device__ inline v8f wmma_bf16(v16bf a, v16bf b, v8f c){
  return __builtin_amdgcn_wmma_f32_16x16x32_bf16(false, a, false, b, (short)0, c,
                                                 false, false);
}

// --- K0: spatial mean of x ---------------------------------------------------
__global__ void xmean_kernel(const float* __restrict__ x, float* __restrict__ xmean){
  __shared__ float red[256];
  int row = blockIdx.x;                       // b*256 + c
  const float* p = x + (size_t)row * 4096;
  float s = 0.f;
  for (int i = threadIdx.x; i < 4096; i += 256) s += p[i];
  red[threadIdx.x] = s; __syncthreads();
  for (int st = 128; st > 0; st >>= 1){
    if (threadIdx.x < st) red[threadIdx.x] += red[threadIdx.x + st];
    __syncthreads();
  }
  if (threadIdx.x == 0) xmean[row] = red[0] * (1.f/4096.f);
}

// --- Kc: fp32 -> bf16 convert -----------------------------------------------
__global__ void f2bf_kernel(const float* __restrict__ in, __bf16* __restrict__ out, int n){
  int i = blockIdx.x * 256 + threadIdx.x;
  if (i < n) out[i] = to_bf(in[i]);
}
// conv2_w [co][ci][3][3] fp32 -> bf16 tap-major [tap][co][ci] (k-contiguous)
__global__ void w2perm_kernel(const float* __restrict__ in, __bf16* __restrict__ out){
  int i = blockIdx.x * 256 + threadIdx.x;
  if (i < 256*256*9){
    int tap = i % 9, rem = i / 9;
    int ci = rem % 256, co = rem / 256;
    out[(size_t)tap*65536 + co*256 + ci] = to_bf(in[i]);
  }
}

// --- K1: g = W1 @ (x - xmean), bf16 out.  Tile 128(co) x 64(n), K=256 -------
__global__ void conv1_center_kernel(const float* __restrict__ x,
                                    const float* __restrict__ xmean,
                                    const __bf16* __restrict__ w1bf,
                                    __bf16* __restrict__ gbf){
  __shared__ __align__(16) __bf16 As[128*32];
  __shared__ __align__(16) __bf16 BsT[64*32];
  int b   = blockIdx.z;
  int co0 = blockIdx.y * 128;
  int n0  = blockIdx.x * 64;
  int tid = threadIdx.x, lane = tid & 31, wv = tid >> 5;
  int mw = (wv >> 1) * 32, nw = (wv & 1) * 32;
  v8f zero = {};
  v8f acc[2][2];
  for (int i = 0; i < 2; ++i) for (int j = 0; j < 2; ++j) acc[i][j] = zero;

  for (int kk = 0; kk < 256; kk += 32){
    // A tile: k-contiguous in global -> async straight into LDS
    for (int e = tid; e < 512; e += 256){
      int m = e >> 2, kq = e & 3;
      async_b128(&As[m*32 + kq*8], w1bf + (size_t)(co0 + m)*256 + kk + kq*8);
    }
    // B tile (centered x) computed in VALU, stored fragment-major BsT[n][k]
    for (int e = tid; e < 256; e += 256){
      int n = e >> 2, kq = e & 3;
      v8bf v;
      #pragma unroll
      for (int j = 0; j < 8; ++j){
        int ci = kk + kq*8 + j;
        v[j] = to_bf(x[((size_t)(b*256 + ci))*4096 + n0 + n] - xmean[b*256 + ci]);
      }
      *(v8bf*)&BsT[n*32 + kq*8] = v;
    }
    wait_async();
    __syncthreads();
    #pragma unroll
    for (int mt = 0; mt < 2; ++mt){
      v16bf a = load_A_frag(&As[(mw + mt*16)*32], 32, lane);
      #pragma unroll
      for (int nt = 0; nt < 2; ++nt){
        v16bf bb = load_B_fragT(&BsT[(nw + nt*16)*32], lane);
        acc[mt][nt] = wmma_bf16(a, bb, acc[mt][nt]);
      }
    }
    __syncthreads();
  }
  int nn = lane & 15, half = lane >> 4;
  for (int mt = 0; mt < 2; ++mt)
    for (int nt = 0; nt < 2; ++nt)
      #pragma unroll
      for (int r = 0; r < 8; ++r){
        int co = co0 + mw + mt*16 + r + half*8;
        int n  = n0 + nw + nt*16 + nn;
        gbf[((size_t)(b*256 + co))*4096 + n] = to_bf(acc[mt][nt][r]);
      }
}

// --- K2: cmat = g g^T / B, bf16 out.  Block 128x128, K = 4096, fully async --
__global__ void cov_kernel(const __bf16* __restrict__ gbf, __bf16* __restrict__ cmatbf){
  __shared__ __align__(16) __bf16 As [128*32];
  __shared__ __align__(16) __bf16 BsT[128*32];
  int b  = blockIdx.z;
  int r0 = blockIdx.x * 128, c0 = blockIdx.y * 128;
  int tid = threadIdx.x, lane = tid & 31, wv = tid >> 5;
  v8f zero = {};
  v8f acc[8];
  for (int i = 0; i < 8; ++i) acc[i] = zero;
  const __bf16* gB = gbf + (size_t)b*256*4096;

  for (int nn = 0; nn < 4096; nn += 32){
    if ((tid & 63) == 0)
      __builtin_prefetch(gB + (size_t)r0*4096 + nn + 256, 0, 1);
    // both tiles are k-contiguous rows of g -> pure async global->LDS staging
    for (int e = tid; e < 512; e += 256){
      int m = e >> 2, kq = e & 3;
      async_b128(&As[m*32 + kq*8], gB + (size_t)(r0 + m)*4096 + nn + kq*8);
    }
    for (int e = tid; e < 512; e += 256){
      int c = e >> 2, kq = e & 3;
      async_b128(&BsT[c*32 + kq*8], gB + (size_t)(c0 + c)*4096 + nn + kq*8);
    }
    wait_async();
    __syncthreads();
    v16bf a = load_A_frag(&As[(wv*16)*32], 32, lane);
    #pragma unroll
    for (int nt = 0; nt < 8; ++nt){
      v16bf bb = load_B_fragT(&BsT[(nt*16)*32], lane);
      acc[nt] = wmma_bf16(a, bb, acc[nt]);
    }
    __syncthreads();
  }
  int n2 = lane & 15, half = lane >> 4;
  for (int nt = 0; nt < 8; ++nt)
    for (int r = 0; r < 8; ++r){
      int rr = r0 + wv*16 + r + half*8;
      int cc = c0 + nt*16 + n2;
      cmatbf[((size_t)b*256 + rr)*256 + cc] = to_bf(acc[nt][r] * (1.0f/16.0f));
    }
}

// --- K3: top-16 eigvecs by blocked subspace iteration (one WG per batch) ----
// LDS: cbf 128KB + Q 16KB + Znew 16KB + QbfT 8KB + red 1KB = 169KB (<320KB WGP)
__global__ __launch_bounds__(256) void eig_kernel(const __bf16* __restrict__ cmbf,
                                                  float* __restrict__ wvec){
  extern __shared__ __align__(16) char sm[];
  __bf16* cbf  = (__bf16*)sm;                           // [256][256]
  float*  Q    = (float*)(sm + 131072);                 // [256][16]
  float*  Znew = (float*)(sm + 131072 + 16384);         // [256][16]
  __bf16* QbfT = (__bf16*)(sm + 131072 + 32768);        // [16][256] (k-contig)
  float*  red  = (float*)(sm + 131072 + 32768 + 8192);  // [256]
  int b = blockIdx.x, tid = threadIdx.x, lane = tid & 31, wv = tid >> 5;
  const __bf16* C = cmbf + (size_t)b*256*256;

  // ---- bulk-load the whole 128KB covariance tile into LDS via the TDM -----
#if __has_builtin(__builtin_amdgcn_tensor_load_to_lds)
  if (wv == 0){
    unsigned long long ga = (unsigned long long)(uintptr_t)C;
    u32x4 g0; i32x8 g1;
    i32x4 g2; i32x8 g4;
    for (int i = 0; i < 4; ++i) g2[i] = 0;
    for (int i = 0; i < 8; ++i){ g1[i] = 0; g4[i] = 0; }
    g0[0] = 1u;                                      // count=1 (valid user D#)
    g0[1] = (unsigned)(uintptr_t)cbf;                // lds_addr (bytes)
    g0[2] = (unsigned)(ga & 0xFFFFFFFFu);            // global_addr[31:0]
    g0[3] = (unsigned)((ga >> 32) & 0x01FFFFFFu) | (2u << 30); // addr hi | type=2
    g1[0] = (1 << 16);                               // data_size = 2 bytes
    g1[1] = (int)(256u << 16);                       // tensor_dim0 = 256 (lo16)
    g1[2] = (int)(256u << 16);                       // tensor_dim1 = 256 (lo16)
    g1[3] = (int)(256u << 16);                       // tile_dim0 = 256
    g1[4] = 256;                                     // tile_dim1 = 256
    g1[5] = 256;                                     // tensor_dim0_stride = 256
#if __clang_major__ >= 23
    __builtin_amdgcn_tensor_load_to_lds(g0, g1, g2, g2, g4, 0);
#else
    __builtin_amdgcn_tensor_load_to_lds(g0, g1, g2, g2, 0);
#endif
    __builtin_amdgcn_s_wait_tensorcnt(0);
  }
#else
  for (int e = tid; e < 256*256; e += 256) cbf[e] = C[e];
#endif
  for (int e = tid; e < 256*16; e += 256){
    unsigned h = (unsigned)(e + 1) * 2654435761u; h ^= h >> 13; h *= 0x85ebca6bu; h ^= h >> 16;
    Q[e] = ((float)(h & 0xFFFF) * (1.f/65536.f)) - 0.5f;
  }
  __syncthreads();

  for (int it = 0; it < 32; ++it){
    for (int e = tid; e < 256*16; e += 256){
      int n = e >> 8, k = e & 255;
      QbfT[n*256 + k] = to_bf(Q[k*16 + n]);          // fragment-major transpose
    }
    __syncthreads();
    // Z = C @ Q : each wave handles 32 rows (2 M-tiles), single N-tile of 16
    v8f zero = {};
    v8f acc[2]; acc[0] = zero; acc[1] = zero;
    for (int kk = 0; kk < 256; kk += 32){
      v16bf bb = load_B_fragT_ld(QbfT + kk, 256, lane);
      #pragma unroll
      for (int mt = 0; mt < 2; ++mt){
        v16bf a = load_A_frag(&cbf[(wv*32 + mt*16)*256 + kk], 256, lane);
        acc[mt] = wmma_bf16(a, bb, acc[mt]);
      }
    }
    int nn = lane & 15, half = lane >> 4;
    for (int mt = 0; mt < 2; ++mt)
      for (int r = 0; r < 8; ++r)
        Znew[(wv*32 + mt*16 + r + half*8)*16 + nn] = acc[mt][r];
    __syncthreads();
    // Modified Gram-Schmidt (thread tid owns row tid)
    int rr = tid;
    for (int j = 0; j < 16; ++j){
      float zj = Znew[rr*16 + j];
      red[tid] = zj*zj; __syncthreads();
      for (int s = 128; s > 0; s >>= 1){
        if (tid < s) red[tid] += red[tid + s];
        __syncthreads();
      }
      float inv = rsqrtf(red[0] + 1e-20f); __syncthreads();
      float qj = zj * inv;
      Q[rr*16 + j] = qj;
      for (int k2 = j + 1; k2 < 16; ++k2){
        red[tid] = qj * Znew[rr*16 + k2]; __syncthreads();
        for (int s = 128; s > 0; s >>= 1){
          if (tid < s) red[tid] += red[tid + s];
          __syncthreads();
        }
        float d = red[0]; __syncthreads();
        Znew[rr*16 + k2] -= d * qj;
      }
    }
    __syncthreads();
  }
  for (int e = tid; e < 16*256; e += 256){
    int k = e >> 8, c = e & 255;
    wvec[((size_t)b*16 + k)*256 + c] = Q[c*16 + k];
  }
}

// --- K4: proj (WMMA) -> softmax(16) -> y = W^T z (WMMA, K padded to 32) -----
__global__ void attn_kernel(const __bf16* __restrict__ gbf,
                            const float* __restrict__ wvec,
                            __bf16* __restrict__ ybf){
  __shared__ __align__(16) __bf16 Wbf [16*256];  // W[k][c]
  __shared__ __align__(16) __bf16 WT  [256*32];  // W^T padded: [c][k], k>=16 -> 0
  __shared__ __align__(16) __bf16 BsT [64*32];
  __shared__ float  Pz[16*64];
  __shared__ __align__(16) __bf16 ZbfT[64*32];   // z fragment-major [n][k]
  int b = blockIdx.y, n0 = blockIdx.x * 64;
  int tid = threadIdx.x, lane = tid & 31, wv = tid >> 5;
  const float*  W  = wvec + (size_t)b*16*256;
  const __bf16* gB = gbf  + (size_t)b*256*4096;
  for (int e = tid; e < 16*256; e += 256) Wbf[e] = to_bf(W[e]);
  __syncthreads();
  for (int e = tid; e < 256*32; e += 256){
    int c = e >> 5, k = e & 31;
    WT[e] = (k < 16) ? Wbf[k*256 + c] : to_bf(0.f);
  }
  v8f zero = {};

  // proj: P[16x64] = W[16x256] @ g[256x64]; waves 0..3 own one 16-col n-tile
  v8f accp = zero;
  for (int kk = 0; kk < 256; kk += 32){
    for (int e = tid; e < 256; e += 256){
      int n = e >> 2, kq = e & 3;
      v8bf v;
      #pragma unroll
      for (int j = 0; j < 8; ++j)
        v[j] = gB[(size_t)(kk + kq*8 + j)*4096 + n0 + n];
      *(v8bf*)&BsT[n*32 + kq*8] = v;
    }
    __syncthreads();
    if (wv < 4){
      v16bf a  = load_A_frag(&Wbf[kk], 256, lane);
      v16bf bb = load_B_fragT(&BsT[(wv*16)*32], lane);
      accp = wmma_bf16(a, bb, accp);
    }
    __syncthreads();
  }
  if (wv < 4){
    int nn = lane & 15, half = lane >> 4;
    for (int r = 0; r < 8; ++r)
      Pz[(r + half*8)*64 + wv*16 + nn] = accp[r] * 3.0f;   // SCALE
  }
  __syncthreads();

  // softmax across the 16 components, per column; K-pad rows 16..31 to zero
  if (tid < 64){
    int col = tid;
    float mx = -1e30f;
    for (int j = 0; j < 16; ++j) mx = fmaxf(mx, Pz[j*64 + col]);
    float ex[16], s = 0.f;
    for (int j = 0; j < 16; ++j){ ex[j] = __expf(Pz[j*64 + col] - mx); s += ex[j]; }
    float inv = 1.f / s;
    for (int j = 0; j < 16; ++j)  ZbfT[col*32 + j] = to_bf(ex[j] * inv);
    for (int j = 16; j < 32; ++j) ZbfT[col*32 + j] = to_bf(0.f);
  }
  __syncthreads();

  // y[256x64] = W^T[256x16] @ z[16x64]; wave wv owns rows [wv*32, wv*32+32)
  v8f acc[2][4];
  for (int i = 0; i < 2; ++i) for (int j = 0; j < 4; ++j) acc[i][j] = zero;
  for (int mt = 0; mt < 2; ++mt){
    v16bf a = load_A_frag(&WT[(wv*32 + mt*16)*32], 32, lane);
    #pragma unroll
    for (int nt = 0; nt < 4; ++nt){
      v16bf bb = load_B_fragT(&ZbfT[(nt*16)*32], lane);
      acc[mt][nt] = wmma_bf16(a, bb, acc[mt][nt]);
    }
  }
  int nn = lane & 15, half = lane >> 4;
  for (int mt = 0; mt < 2; ++mt)
    for (int nt = 0; nt < 4; ++nt)
      for (int r = 0; r < 8; ++r){
        int c = wv*32 + mt*16 + r + half*8;
        ybf[((size_t)(b*256 + c))*4096 + n0 + nt*16 + nn] = to_bf(acc[mt][nt][r]);
      }
}

// --- K5: 3x3 conv (implicit GEMM) + BN affine + residual --------------------
__global__ void conv2_bn_res_kernel(const float* __restrict__ x,
                                    const __bf16* __restrict__ ybf,
                                    const __bf16* __restrict__ w2t,
                                    const float* __restrict__ gamma,
                                    const float* __restrict__ beta,
                                    const float* __restrict__ mean,
                                    const float* __restrict__ var,
                                    float* __restrict__ out){
  __shared__ __align__(16) __bf16 Ys[32][10][10];   // 32 ci, 8x8 tile + halo
  __shared__ __align__(16) __bf16 As [128*32];
  __shared__ __align__(16) __bf16 BsT[64*32];
  int b = blockIdx.z, co0 = blockIdx.y * 128;
  int tile = blockIdx.x;                            // 64 spatial tiles of 8x8
  int h0 = (tile >> 3) * 8, w0 = (tile & 7) * 8;
  int tid = threadIdx.x, lane = tid & 31, wv = tid >> 5;
  int mw = (wv >> 1) * 32, nw = (wv & 1) * 32;
  v8f zero = {};
  v8f acc[2][2];
  for (int i = 0; i < 2; ++i) for (int j = 0; j < 2; ++j) acc[i][j] = zero;
  const __bf16* yB = ybf + (size_t)b*256*4096;

  for (int cc = 0; cc < 256; cc += 32){
    for (int e = tid; e < 32*100; e += 256){
      int k = e / 100, rem = e % 100, hh = rem / 10, ww = rem % 10;
      int hg = h0 + hh - 1, wg = w0 + ww - 1;
      __bf16 v = to_bf(0.f);
      if (hg >= 0 && hg < 64 && wg >= 0 && wg < 64)
        v = yB[(size_t)(cc + k)*4096 + hg*64 + wg];
      Ys[k][hh][ww] = v;
    }
    __syncthreads();
    for (int tap = 0; tap < 9; ++tap){
      int dy = tap / 3, dx = tap % 3;
      // A tile from tap-major weights: k-contiguous -> async global->LDS
      const __bf16* wt = w2t + (size_t)tap*65536;
      for (int e = tid; e < 512; e += 256){
        int mm = e >> 2, kq = e & 3;
        async_b128(&As[mm*32 + kq*8], wt + (size_t)(co0 + mm)*256 + cc + kq*8);
      }
      // B tile: shifted window gathered once into fragment-major BsT
      for (int e = tid; e < 2048; e += 256){
        int n = e >> 5, k = e & 31;
        int hh = n >> 3, ww = n & 7;
        BsT[e] = Ys[k][hh + dy][ww + dx];
      }
      wait_async();
      __syncthreads();
      #pragma unroll
      for (int mt = 0; mt < 2; ++mt){
        v16bf a = load_A_frag(&As[(mw + mt*16)*32], 32, lane);
        #pragma unroll
        for (int nt = 0; nt < 2; ++nt){
          v16bf bb = load_B_fragT(&BsT[(nw + nt*16)*32], lane);
          acc[mt][nt] = wmma_bf16(a, bb, acc[mt][nt]);
        }
      }
      __syncthreads();
    }
  }
  // fused BN (eval affine) + residual
  int n = lane & 15, half = lane >> 4;
  for (int mt = 0; mt < 2; ++mt)
    for (int nt = 0; nt < 2; ++nt)
      for (int r = 0; r < 8; ++r){
        int co = co0 + mw + mt*16 + r + half*8;
        int nl = nw + nt*16 + n;
        int h = h0 + (nl >> 3), wc = w0 + (nl & 7);
        float inv = gamma[co] * rsqrtf(var[co] + 1e-5f);
        float bt  = beta[co] - mean[co]*inv;
        size_t idx = (((size_t)b*256 + co)*64 + h)*64 + wc;
        out[idx] = x[idx] + acc[mt][nt][r]*inv + bt;
      }
}

// ---------------------------------------------------------------------------
extern "C" void kernel_launch(void* const* d_in, const int* in_sizes, int n_in,
                              void* d_out, int out_size, void* d_ws, size_t ws_size,
                              hipStream_t stream){
  const float* x   = (const float*)d_in[0];
  const float* w1  = (const float*)d_in[1];
  // d_in[2] = conv1_b: cancels exactly under mean-centering; unused.
  const float* w2  = (const float*)d_in[3];
  const float* gam = (const float*)d_in[4];
  const float* bet = (const float*)d_in[5];
  const float* mu  = (const float*)d_in[6];
  const float* vr  = (const float*)d_in[7];
  float* out = (float*)d_out;

  char* ws = (char*)d_ws;
  size_t off = 0;
  __bf16* gbf  = (__bf16*)(ws + off); off += (size_t)16*256*4096*2;  // 32 MB
  __bf16* ybf  = (__bf16*)(ws + off); off += (size_t)16*256*4096*2;  // 32 MB
  __bf16* w1bf = (__bf16*)(ws + off); off += (size_t)256*256*2;
  __bf16* w2t  = (__bf16*)(ws + off); off += (size_t)9*256*256*2;    // tap-major
  float* xmean = (float*)(ws + off); off += (size_t)16*256*4;
  __bf16* cmbf = (__bf16*)(ws + off); off += (size_t)16*256*256*2;   // bf16 cov
  float* wvec  = (float*)(ws + off); off += (size_t)16*16*256*4;

  hipLaunchKernelGGL(xmean_kernel, dim3(16*256), dim3(256), 0, stream, x, xmean);
  hipLaunchKernelGGL(f2bf_kernel, dim3((256*256 + 255)/256), dim3(256), 0, stream,
                     w1, w1bf, 256*256);
  hipLaunchKernelGGL(w2perm_kernel, dim3((256*256*9 + 255)/256), dim3(256), 0, stream,
                     w2, w2t);
  hipLaunchKernelGGL(conv1_center_kernel, dim3(64, 2, 16), dim3(256), 0, stream,
                     x, xmean, w1bf, gbf);
  hipLaunchKernelGGL(cov_kernel, dim3(2, 2, 16), dim3(256), 0, stream, gbf, cmbf);
  size_t eig_lds = 131072 + 16384 + 16384 + 8192 + 1024;  // 169 KB of 320 KB WGP LDS
  hipLaunchKernelGGL(eig_kernel, dim3(16), dim3(256), eig_lds, stream, cmbf, wvec);
  hipLaunchKernelGGL(attn_kernel, dim3(64, 16), dim3(256), 0, stream, gbf, wvec, ybf);
  hipLaunchKernelGGL(conv2_bn_res_kernel, dim3(64, 2, 16), dim3(256), 0, stream,
                     x, ybf, w2t, gam, bet, mu, vr, out);
  (void)in_sizes; (void)n_in; (void)out_size; (void)ws_size;
}
